// LSTM_seq2seq_6897717477681
// MI455X (gfx1250) — matrix operand order repaired and checked by reading, not compiled, not run on hardware
//
#include <hip/hip_runtime.h>

#define HS    520      // LDS h row stride in halves (512 + 8 pad -> conflict-free b128)
#define WGS   512      // 16 wave32 per workgroup
#define BM    16       // batch rows per workgroup
#define HSZ   512
#define G4    2048
#define TSRC  50
#define TTGT  30
#define ID    4

typedef __attribute__((ext_vector_type(16))) _Float16 v16h;
typedef __attribute__((ext_vector_type(8)))  _Float16 v8h;
typedef __attribute__((ext_vector_type(8)))  float    v8f;

union V16U { v16h v; v8h h[2]; };

__device__ __forceinline__ v8f zero8() {
    v8f a;
#pragma unroll
    for (int i = 0; i < 8; i++) a[i] = 0.f;
    return a;
}

__device__ __forceinline__ float sigmoidf_(float x) { return 1.f / (1.f + __expf(-x)); }
// stable tanh via exp: 1 - 2/(e^{2x}+1)
__device__ __forceinline__ float tanhf_(float x) { float e = __expf(2.f * x); return 1.f - 2.f / (e + 1.f); }

// A fragment: 16x32 f16 tile of h (row-major LDS, stride HS).
// lane<16:  row=lane,    K = {k0..k0+7, k0+16..k0+23}
// lane>=16: row=lane-16, K = {k0+8..k0+15, k0+24..k0+31}
__device__ __forceinline__ v16h load_a_frag(const _Float16* A, int lane, int k0) {
    const int row  = lane & 15;
    const int koff = (lane < 16) ? 0 : 8;
    V16U u;
    u.h[0] = *(const v8h*)(A + row * HS + k0 + koff);
    u.h[1] = *(const v8h*)(A + row * HS + k0 + 16 + koff);
    return u.v;
}

// B fragment: 32x16 tile of W^T == W[n..n+15][k0..k0+31], W row-major (2048x512) f16.
// lane<16: col=lane, K={k0..k0+7, k0+16..k0+23} ; lane>=16: col=lane-16, K={k0+8..k0+15, k0+24..k0+31}
__device__ __forceinline__ v16h load_b_frag(const _Float16* __restrict__ W, int lane, int nbase, int k0) {
    const int col  = lane & 15;
    const int koff = (lane < 16) ? 0 : 8;
    const _Float16* p = W + (((size_t)(nbase + col)) << 9) + k0 + koff;
    V16U u;
    u.h[0] = *(const v8h*)(p);
    u.h[1] = *(const v8h*)(p + 16);
    return u.v;
}

// One matmul chain: acc[u] += A(16x512) x W^T slice, tiles paired to share A-frags
// and to keep only ~2 independent WMMA chains + a few fragments live at once.
__device__ __forceinline__ void mm_chain(const _Float16* A,
                                         const _Float16* __restrict__ W,
                                         int lane, int wbase, v8f* acc) {
#pragma unroll
    for (int p = 0; p < 4; p++) {            // p = quadrant; tiles u0=2p (t=0), u1=2p+1 (t=1)
        const int nb0 = p * HSZ + wbase;
        const int nb1 = nb0 + 16;
        v8f a0 = acc[2 * p], a1 = acc[2 * p + 1];
#pragma unroll 2
        for (int k0 = 0; k0 < HSZ; k0 += 32) {
            const v16h av  = load_a_frag(A, lane, k0);
            const v16h bv0 = load_b_frag(W, lane, nb0, k0);
            const v16h bv1 = load_b_frag(W, lane, nb1, k0);
            a0 = __builtin_amdgcn_wmma_f32_16x16x32_f16(false, av, false, bv0, (short)0, a0, false, false);
            a1 = __builtin_amdgcn_wmma_f32_16x16x32_f16(false, av, false, bv1, (short)0, a1, false, false);
        }
        acc[2 * p] = a0; acc[2 * p + 1] = a1;
    }
}

// One LSTM cell step for BM=16 rows. Wave owns hidden cols [wave*32, wave*32+32)
// in all four gate quadrants; c state stays in registers (WMMA C layout).
template <bool HAS_X, bool HAS_IH>
__device__ __forceinline__ void lstm_cell(
    int lane, int wbase,
    const _Float16* Aih,                     // LDS h feeding the Wih chain (layer1)
    const _Float16* Ahh,                     // LDS h feeding the Whh chain
    const _Float16* __restrict__ Wih16,      // f16 (2048x512) or unused
    const _Float16* __restrict__ Whh16,      // f16 (2048x512)
    const float* __restrict__ bias,          // 2048
    const float* __restrict__ Wih_x,         // f32 (2048x4) when HAS_X
    const float* xb,                         // LDS 16x4 when HAS_X
    v8f* cfrag,                              // [2] in/out
    _Float16* hout)                          // LDS h output (stride HS)
{
    const int col    = lane & 15;
    const int rowoff = (lane < 16) ? 0 : 8;

    // init accumulators: bias (+ tiny I=4 input GEMM done on VALU)
    v8f acc[8];
#pragma unroll
    for (int q = 0; q < 4; q++) {
#pragma unroll
        for (int t = 0; t < 2; t++) {
            const int n = q * HSZ + wbase + t * 16 + col;
            const float bb = bias[n];
            v8f a;
            if (HAS_X) {
                const float w0 = Wih_x[n * 4 + 0], w1 = Wih_x[n * 4 + 1];
                const float w2 = Wih_x[n * 4 + 2], w3 = Wih_x[n * 4 + 3];
#pragma unroll
                for (int r = 0; r < 8; r++) {
                    const float* xr = xb + (r + rowoff) * 4;
                    a[r] = bb + xr[0] * w0 + xr[1] * w1 + xr[2] * w2 + xr[3] * w3;
                }
            } else {
#pragma unroll
                for (int r = 0; r < 8; r++) a[r] = bb;
            }
            acc[q * 2 + t] = a;
        }
    }

    if (HAS_IH) mm_chain(Aih, Wih16, lane, wbase, acc);   // gates += h_in  @ Wih^T
    mm_chain(Ahh, Whh16, lane, wbase, acc);               // gates += h_prev @ Whh^T

    // pointwise LSTM directly on accumulators (i,f,g,o share lane/VGPR positions)
    _Float16 hval[2][8];
#pragma unroll
    for (int t = 0; t < 2; t++) {
        const v8f ai = acc[t], af = acc[2 + t], ag = acc[4 + t], ao = acc[6 + t];
        v8f c = cfrag[t];
#pragma unroll
        for (int r = 0; r < 8; r++) {
            const float cc = sigmoidf_(af[r]) * c[r] + sigmoidf_(ai[r]) * tanhf_(ag[r]);
            c[r] = cc;
            hval[t][r] = (_Float16)(sigmoidf_(ao[r]) * tanhf_(cc));
        }
        cfrag[t] = c;
    }

    __syncthreads();   // everyone finished reading old h
#pragma unroll
    for (int t = 0; t < 2; t++)
#pragma unroll
        for (int r = 0; r < 8; r++)
            hout[(r + rowoff) * HS + wbase + t * 16 + col] = hval[t][r];
    __syncthreads();   // new h visible
}

__global__ void __launch_bounds__(WGS, 1) lstm_seq2seq_kernel(
    const float* __restrict__ x,
    const float* __restrict__ enc_Wih0, const float* __restrict__ enc_b0,
    const float* __restrict__ enc_b1,
    const float* __restrict__ dec_Wih0, const float* __restrict__ dec_b0,
    const float* __restrict__ dec_b1,
    const float* __restrict__ head_W, const float* __restrict__ head_b,
    const _Float16* __restrict__ w16,
    float* __restrict__ out)
{
    __shared__ _Float16 h0buf[BM * HS];
    __shared__ _Float16 h1buf[BM * HS];
    __shared__ float    xbuf[BM * ID];

    const int tid   = threadIdx.x;
    const int lane  = tid & 31;
    const int wave  = tid >> 5;
    const int wbase = wave * 32;             // 32 hidden cols per wave, 16 waves
    const int bbase = blockIdx.x * BM;

    const size_t MSZ = (size_t)G4 * HSZ;
    const _Float16* eWhh0 = w16 + 0 * MSZ;
    const _Float16* eWih1 = w16 + 1 * MSZ;
    const _Float16* eWhh1 = w16 + 2 * MSZ;
    const _Float16* dWhh0 = w16 + 3 * MSZ;
    const _Float16* dWih1 = w16 + 4 * MSZ;
    const _Float16* dWhh1 = w16 + 5 * MSZ;

    for (int i = tid; i < BM * HS; i += WGS) { h0buf[i] = (_Float16)0.f; h1buf[i] = (_Float16)0.f; }
    v8f c0[2], c1[2];
#pragma unroll
    for (int t = 0; t < 2; t++) { c0[t] = zero8(); c1[t] = zero8(); }
    __syncthreads();

    // ---------------- encoder ----------------
    for (int ts = 0; ts < TSRC; ts++) {
        if (tid < BM * ID) {
            const int r = tid >> 2, k = tid & 3;
            xbuf[tid] = x[((size_t)(bbase + r) * TSRC + ts) * ID + k];
        }
        __syncthreads();
        lstm_cell<true,  false>(lane, wbase, h0buf, h0buf, eWih1, eWhh0, enc_b0,
                                enc_Wih0, xbuf, c0, h0buf);
        lstm_cell<false, true >(lane, wbase, h0buf, h1buf, eWih1, eWhh1, enc_b1,
                                nullptr, nullptr, c1, h1buf);
    }

    // decoder initial input = x[:, -1, :]
    if (tid < BM * ID) {
        const int r = tid >> 2, k = tid & 3;
        xbuf[tid] = x[((size_t)(bbase + r) * TSRC + (TSRC - 1)) * ID + k];
    }
    __syncthreads();

    // ---------------- decoder ----------------
    for (int ts = 0; ts < TTGT; ts++) {
        lstm_cell<true,  false>(lane, wbase, h0buf, h0buf, dWih1, dWhh0, dec_b0,
                                dec_Wih0, xbuf, c0, h0buf);
        lstm_cell<false, true >(lane, wbase, h0buf, h1buf, dWih1, dWhh1, dec_b1,
                                nullptr, nullptr, c1, h1buf);
        // head: out = h1 @ head_W^T + head_b  (16x512 · 512x4, VALU)
        if (tid < BM * ID) {
            const int r = tid >> 2, j = tid & 3;
            const float* wr = head_W + j * HSZ;
            float s = head_b[j];
            for (int k = 0; k < HSZ; k++) s += (float)h1buf[r * HS + k] * wr[k];
            out[((size_t)(bbase + r) * TTGT + ts) * ID + j] = s;
            xbuf[tid] = s;   // feed back as next decoder input
        }
        __syncthreads();
    }
}

__global__ void cvt_f32_to_f16(const float* __restrict__ src, _Float16* __restrict__ dst, int n) {
    const int i = blockIdx.x * blockDim.x + threadIdx.x;
    if (i < n) dst[i] = (_Float16)src[i];
}

extern "C" void kernel_launch(void* const* d_in, const int* in_sizes, int n_in,
                              void* d_out, int out_size, void* d_ws, size_t ws_size,
                              hipStream_t stream) {
    const float* x      = (const float*)d_in[0];
    const float* eWih0  = (const float*)d_in[1];
    const float* eWhh0  = (const float*)d_in[2];
    const float* eb0    = (const float*)d_in[3];
    const float* eWih1  = (const float*)d_in[4];
    const float* eWhh1  = (const float*)d_in[5];
    const float* eb1    = (const float*)d_in[6];
    const float* dWih0  = (const float*)d_in[7];
    const float* dWhh0  = (const float*)d_in[8];
    const float* db0    = (const float*)d_in[9];
    const float* dWih1  = (const float*)d_in[10];
    const float* dWhh1  = (const float*)d_in[11];
    const float* db1    = (const float*)d_in[12];
    const float* headW  = (const float*)d_in[13];
    const float* headb  = (const float*)d_in[14];
    (void)in_sizes; (void)n_in; (void)out_size; (void)ws_size;

    _Float16* w16 = (_Float16*)d_ws;
    const size_t M = (size_t)G4 * HSZ;     // 1,048,576 elems per matrix
    const float* srcs[6] = {eWhh0, eWih1, eWhh1, dWhh0, dWih1, dWhh1};
    for (int i = 0; i < 6; i++)
        cvt_f32_to_f16<<<(int)((M + 255) / 256), 256, 0, stream>>>(srcs[i], w16 + (size_t)i * M, (int)M);

    lstm_seq2seq_kernel<<<1024 / BM, WGS, 0, stream>>>(
        x, eWih0, eb0, eb1, dWih0, db0, db1, headW, headb, w16, (float*)d_out);
}
